// YKT_68753836474920
// MI455X (gfx1250) — matrix operand order repaired and checked by reading, not compiled, not run on hardware
//
#include <hip/hip_runtime.h>
#include <hip/hip_bf16.h>

// MI455X / gfx1250, wave32. f16 WMMA (f32 accum) for all matmuls, fp32 softmax/LN.
// Weights/activations pre-converted to f16 once; GEMM tiles staged via TDM
// (tensor_load_to_lds, double-buffered, s_wait_tensorcnt) when available.

#define B_  16
#define S_  512
#define D_  512
#define H_  8
#define HD_ 64
#define NEGV (-1.0e9f)

typedef __attribute__((ext_vector_type(16))) _Float16 v16h;
typedef __attribute__((ext_vector_type(8)))  float    v8f;
typedef __attribute__((ext_vector_type(4))) unsigned int u32x4_t;
typedef __attribute__((ext_vector_type(4))) int          i32x4_t;
typedef __attribute__((ext_vector_type(8))) int          i32x8_t;

#define AS1_ __attribute__((address_space(1)))
#define AS3_ __attribute__((address_space(3)))

#if defined(__HIP_DEVICE_COMPILE__) && __has_builtin(__builtin_amdgcn_tensor_load_to_lds) && __has_builtin(__builtin_amdgcn_s_wait_tensorcnt)
#define HAVE_TDM 1
#else
#define HAVE_TDM 0
#endif

#if defined(__HIP_DEVICE_COMPILE__) && __has_builtin(__builtin_amdgcn_global_load_async_to_lds_b128) && __has_builtin(__builtin_amdgcn_s_wait_asynccnt)
#define HAVE_ASYNC 1
#else
#define HAVE_ASYNC 0
#endif

__device__ __forceinline__ v8f wmma16(v16h a, v16h b, v8f c) {
  return __builtin_amdgcn_wmma_f32_16x16x32_f16(false, a, false, b, (short)0, c, false, false);
}

// A fragment: 16(M) x 32(K) f16 row-major tile, 'stride' halves per row.
__device__ __forceinline__ v16h load_frag_a16(const _Float16* base, int stride) {
  const int lane = threadIdx.x & 31;
  const _Float16* p = base + (lane & 15) * stride + ((lane >> 4) << 3);
  union { v16h v; uint4 q[2]; } u;
  u.q[0] = *(const uint4*)(p);
  u.q[1] = *(const uint4*)(p + 16);
  return u.v;
}
// B fragment from 16(N) x 32(K) row-major tile: B[k][n] = base[n*stride + k].
__device__ __forceinline__ v16h load_frag_b16(const _Float16* base, int stride) {
  const int lane = threadIdx.x & 31;
  const _Float16* p = base + (lane & 15) * stride + ((lane >> 4) << 3);
  union { v16h v; uint4 q[2]; } u;
  u.q[0] = *(const uint4*)(p);
  u.q[1] = *(const uint4*)(p + 16);
  return u.v;
}

#if HAVE_TDM
__device__ __forceinline__ unsigned lds_off(const void* p) {
  return (unsigned)(size_t)(const AS3_ void*)p;
}

// One TDM descriptor: tile_rows x 32 f16 tile from a [tensor_rows][512] f16
// tensor into LDS with 16B pad per 64B row (-> 40-half LDS row stride).
__device__ __forceinline__ void tdm_load_tile_f16(
    const _Float16* g, unsigned ldsb, int tile_rows, int tensor_rows)
{
  const unsigned long long ga = (unsigned long long)(size_t)g;
  u32x4_t g0;
  g0[0] = 1u;                                           // count=1
  g0[1] = ldsb;                                         // lds_addr
  g0[2] = (unsigned)ga;                                 // global_addr[31:0]
  g0[3] = (unsigned)((ga >> 32) & 0x01FFFFFFu) | (2u << 30);  // addr[56:32] | type=2
  i32x8_t g1;
  g1[0] = (1 << 16)      // data_size = 2 bytes
        | (1 << 20)      // pad_enable
        | (3 << 22)      // pad_interval: 16 DWORDs (=64B, one tile row)
        | (3 << 25);     // pad_amount: 4 DWORDs (=16B)
  g1[1] = (512 & 0xFFFF) << 16;                         // tensor_dim0 lo16
  g1[2] = (512 >> 16) | ((tensor_rows & 0xFFFF) << 16); // dim0 hi / dim1 lo
  g1[3] = (tensor_rows >> 16) | (32 << 16);             // dim1 hi / tile_dim0=32
  g1[4] = tile_rows;                                    // tile_dim1 (tile_dim2=0)
  g1[5] = 512;                                          // tensor_dim0_stride lo32
  g1[6] = 0;
  g1[7] = 0;
  const i32x4_t z4 = {};
#if __clang_major__ >= 23
  const i32x8_t z8 = {};
  __builtin_amdgcn_tensor_load_to_lds(g0, g1, z4, z4, z8, 0);
#else
  __builtin_amdgcn_tensor_load_to_lds(g0, g1, z4, z4, 0);
#endif
}
#endif  // HAVE_TDM

// ---------------------------------------------------------------------------
// K0: f32 -> f16 convert (inputs and the five weight matrices, once).
// ---------------------------------------------------------------------------
__global__ void __launch_bounds__(256) cvt_kernel(
    const float* __restrict__ s, _Float16* __restrict__ d, int n4)
{
  const int i = blockIdx.x * 256 + threadIdx.x;
  if (i >= n4) return;
  const float4 f = ((const float4*)s)[i];
  union { _Float16 h[4]; uint2 u; } pk;
  pk.h[0] = (_Float16)f.x; pk.h[1] = (_Float16)f.y;
  pk.h[2] = (_Float16)f.z; pk.h[3] = (_Float16)f.w;
  ((uint2*)d)[i] = pk.u;
}

// ---------------------------------------------------------------------------
// K2: mem_attn = softmax(log10(ren+1)+exp(-|ts|), causal), head-independent.
// ---------------------------------------------------------------------------
__global__ void __launch_bounds__(256) mem_attn_kernel(
    const float* __restrict__ ren, const float* __restrict__ ts,
    float* __restrict__ Mem)
{
  const int lane = threadIdx.x & 31;
  const int wid  = threadIdx.x >> 5;
  const int row  = blockIdx.x * 8 + wid;
  const int qi   = row & (S_ - 1);
  const size_t base = (size_t)row * S_;
  float vals[16];
  float mx = -3.0e38f;
#pragma unroll
  for (int t = 0; t < 16; ++t) {
    const int c = lane + t * 32;
    float v;
    if (c <= qi) v = __log10f(ren[base + c] + 1.0f) + __expf(-fabsf(ts[base + c]));
    else         v = 2.0f * NEGV;
    vals[t] = v;
    mx = fmaxf(mx, v);
  }
#pragma unroll
  for (int o = 16; o >= 1; o >>= 1) mx = fmaxf(mx, __shfl_xor(mx, o, 32));
  float sum = 0.0f;
#pragma unroll
  for (int t = 0; t < 16; ++t) { float e = __expf(vals[t] - mx); vals[t] = e; sum += e; }
#pragma unroll
  for (int o = 16; o >= 1; o >>= 1) sum += __shfl_xor(sum, o, 32);
  const float rinv = 1.0f / sum;
#pragma unroll
  for (int t = 0; t < 16; ++t) Mem[base + lane + t * 32] = vals[t] * rinv;
}

// ---------------------------------------------------------------------------
// K1: fused QKV projection (all-f16 operands). Block tile 128x64, 8 waves.
// Q,K -> [b][h][s][d] f16 ; V -> transposed [b][h][d][s] f16.
// ---------------------------------------------------------------------------
__global__ void __launch_bounds__(256) qkv_proj_kernel(
    const _Float16* __restrict__ Xh,
    const _Float16* __restrict__ Whq, const _Float16* __restrict__ Whk,
    const _Float16* __restrict__ Whv,
    const float* __restrict__ bq, const float* __restrict__ bk,
    const float* __restrict__ bv,
    _Float16* __restrict__ Qo, _Float16* __restrict__ Ko, _Float16* __restrict__ VTo)
{
  __shared__ _Float16 At[2][128 * 40];
  __shared__ _Float16 Bt[2][3][64 * 40];
  const int tid  = threadIdx.x;
  const int lane = tid & 31;
  const int wid  = tid >> 5;
  const int wm   = wid >> 1;
  const int wn   = wid & 1;
  const int m0   = blockIdx.x * 128;
  const int n0   = blockIdx.y * 64;
  const _Float16* Ws[3] = {Whq, Whk, Whv};

  v8f acc[3][4];
#pragma unroll
  for (int w = 0; w < 3; ++w)
#pragma unroll
    for (int t = 0; t < 4; ++t) acc[w][t] = (v8f){};

#if HAVE_TDM
  if (wid == 0) {
    tdm_load_tile_f16(Xh + (size_t)m0 * D_, lds_off(&At[0][0]), 128, B_ * S_);
#pragma unroll
    for (int w = 0; w < 3; ++w)
      tdm_load_tile_f16(Ws[w] + (size_t)n0 * D_, lds_off(&Bt[0][w][0]), 64, D_);
  }
  for (int kk = 0; kk < 16; ++kk) {
    const int bs = kk & 1;
    if (wid == 0) {
      if (kk < 15) {
        const int k1 = (kk + 1) * 32;
        tdm_load_tile_f16(Xh + (size_t)m0 * D_ + k1, lds_off(&At[bs ^ 1][0]), 128, B_ * S_);
#pragma unroll
        for (int w = 0; w < 3; ++w)
          tdm_load_tile_f16(Ws[w] + (size_t)n0 * D_ + k1, lds_off(&Bt[bs ^ 1][w][0]), 64, D_);
        __builtin_amdgcn_s_wait_tensorcnt(4);
      } else {
        __builtin_amdgcn_s_wait_tensorcnt(0);
      }
    }
    __syncthreads();
    const v16h a0 = load_frag_a16(&At[bs][(wm * 32) * 40], 40);
    const v16h a1 = load_frag_a16(&At[bs][(wm * 32 + 16) * 40], 40);
#pragma unroll
    for (int w = 0; w < 3; ++w) {
      const v16h b0  = load_frag_b16(&Bt[bs][w][(wn * 32) * 40], 40);
      const v16h b1f = load_frag_b16(&Bt[bs][w][(wn * 32 + 16) * 40], 40);
      acc[w][0] = wmma16(a0, b0,  acc[w][0]);
      acc[w][1] = wmma16(a0, b1f, acc[w][1]);
      acc[w][2] = wmma16(a1, b0,  acc[w][2]);
      acc[w][3] = wmma16(a1, b1f, acc[w][3]);
    }
    __syncthreads();
  }
#else
  for (int k0 = 0; k0 < D_; k0 += 32) {
    __syncthreads();
#pragma unroll
    for (int i = 0; i < 2; ++i) {                  // A: 512 uint4
      const int idx = tid + i * 256;
      const int r = idx >> 2, c = (idx & 3) * 8;
      *(uint4*)&At[0][r * 40 + c] = *(const uint4*)(Xh + (size_t)(m0 + r) * D_ + k0 + c);
    }
#pragma unroll
    for (int w = 0; w < 3; ++w) {                  // each W: 256 uint4
      const int r = tid >> 2, c = (tid & 3) * 8;
      *(uint4*)&Bt[0][w][r * 40 + c] = *(const uint4*)(Ws[w] + (size_t)(n0 + r) * D_ + k0 + c);
    }
    __syncthreads();
    const v16h a0 = load_frag_a16(&At[0][(wm * 32) * 40], 40);
    const v16h a1 = load_frag_a16(&At[0][(wm * 32 + 16) * 40], 40);
#pragma unroll
    for (int w = 0; w < 3; ++w) {
      const v16h b0  = load_frag_b16(&Bt[0][w][(wn * 32) * 40], 40);
      const v16h b1f = load_frag_b16(&Bt[0][w][(wn * 32 + 16) * 40], 40);
      acc[w][0] = wmma16(a0, b0,  acc[w][0]);
      acc[w][1] = wmma16(a0, b1f, acc[w][1]);
      acc[w][2] = wmma16(a1, b0,  acc[w][2]);
      acc[w][3] = wmma16(a1, b1f, acc[w][3]);
    }
  }
#endif

  const float* Bs[3] = {bq, bk, bv};
  _Float16* Outs[2] = {Qo, Ko};
#pragma unroll
  for (int w = 0; w < 3; ++w) {
#pragma unroll
    for (int i = 0; i < 2; ++i) {
#pragma unroll
      for (int j = 0; j < 2; ++j) {
        const int n = n0 + wn * 32 + j * 16 + (lane & 15);
        const float bias = Bs[w][n];
        const v8f a = acc[w][i * 2 + j];
        const int hh = n >> 6, d = n & 63;
#pragma unroll
        for (int r = 0; r < 8; ++r) {
          const int m = m0 + wm * 32 + i * 16 + r + ((lane >> 4) << 3);
          const int bi = m >> 9, s = m & (S_ - 1);
          const float val = a[r] + bias;
          if (w < 2) {
            Outs[w][(size_t)(((bi * H_ + hh) * S_) + s) * HD_ + d] = (_Float16)val;
          } else {
            VTo[(size_t)(((bi * H_ + hh) * HD_) + d) * S_ + s] = (_Float16)val;
          }
        }
      }
    }
  }
}

// ---------------------------------------------------------------------------
// K3: attention. Block = (b,h, 16-row q tile), 4 waves.
// ---------------------------------------------------------------------------
__global__ void __launch_bounds__(128) attn_kernel(
    const _Float16* __restrict__ Qf, const _Float16* __restrict__ Kf,
    const _Float16* __restrict__ VT, const float* __restrict__ Mem,
    const float* __restrict__ l1p, _Float16* __restrict__ attnOut)
{
  __shared__ _Float16 Qt[16 * 72];
  __shared__ float    Sc[16 * 513];
  __shared__ _Float16 Pt[16 * 520];
  const int tid  = threadIdx.x;
  const int lane = tid & 31;
  const int wid  = tid >> 5;
  const int q0   = blockIdx.x * 16;
  const int bh   = blockIdx.y;
  const int bi   = bh >> 3;
  const int hh   = bh & 7;
  const _Float16* Qbase = Qf + (size_t)bh * S_ * HD_;
  const _Float16* Kbase = Kf + (size_t)bh * S_ * HD_;

#if HAVE_ASYNC
  {
    const int r = tid >> 3, c = (tid & 7) * 8;      // 128 x b128 = 16x64 halves
    __builtin_amdgcn_global_load_async_to_lds_b128(
        (AS1_ i32x4_t*)(_Float16*)(Qbase + (size_t)(q0 + r) * HD_ + c),
        (AS3_ i32x4_t*)&Qt[r * 72 + c], 0, 0);
    __builtin_amdgcn_s_wait_asynccnt(0);
  }
#else
#pragma unroll
  for (int i = 0; i < 2; ++i) {
    const int idx = tid + i * 128;
    const int r = idx >> 4, c = (idx & 15) * 4;
    *(uint2*)&Qt[r * 72 + c] = *(const uint2*)(Qbase + (size_t)(q0 + r) * HD_ + c);
  }
#endif
  __syncthreads();

  const v16h aq0 = load_frag_a16(Qt, 72);
  const v16h aq1 = load_frag_a16(Qt + 32, 72);

#pragma unroll
  for (int t = 0; t < 8; ++t) {
    const int n0 = wid * 128 + t * 16;
    v8f acc = (v8f){};
    const v16h bk0 = load_frag_b16(Kbase + (size_t)n0 * HD_ +  0, HD_);
    const v16h bk1 = load_frag_b16(Kbase + (size_t)n0 * HD_ + 32, HD_);
    acc = wmma16(aq0, bk0, acc);
    acc = wmma16(aq1, bk1, acc);
#pragma unroll
    for (int r = 0; r < 8; ++r) {
      const int rl  = r + ((lane >> 4) << 3);
      const int col = n0 + (lane & 15);
      Sc[rl * 513 + col] = (col > q0 + rl) ? NEGV : acc[r] * 0.125f;
    }
  }
  __syncthreads();

  {
    const int row = tid >> 3;
    const int cb  = (tid & 7) * 64;
    float* srow = &Sc[row * 513 + cb];
    float mx = -3.0e38f;
    for (int c = 0; c < 64; ++c) mx = fmaxf(mx, srow[c]);
    mx = fmaxf(mx, __shfl_xor(mx, 1, 32));
    mx = fmaxf(mx, __shfl_xor(mx, 2, 32));
    mx = fmaxf(mx, __shfl_xor(mx, 4, 32));
    float sum = 0.0f;
    for (int c = 0; c < 64; ++c) { float e = __expf(srow[c] - mx); srow[c] = e; sum += e; }
    sum += __shfl_xor(sum, 1, 32);
    sum += __shfl_xor(sum, 2, 32);
    sum += __shfl_xor(sum, 4, 32);
    const float lv = l1p[0];
    const float ca = (1.0f - lv) / sum;
    const float* mrow = Mem + ((size_t)bi * S_ + q0 + row) * S_ + cb;
    for (int c = 0; c < 64; ++c)
      Pt[row * 520 + cb + c] = (_Float16)(ca * srow[c] + lv * mrow[c]);
  }
  __syncthreads();

  {
    v8f acc = (v8f){};
    const _Float16* Vbase = VT + ((size_t)bh * HD_ + wid * 16) * S_;
#pragma unroll
    for (int c = 0; c < 16; ++c) {
      const v16h a  = load_frag_a16(Pt + c * 32, 520);
      const v16h bv = load_frag_b16(Vbase + c * 32, S_);
      acc = wmma16(a, bv, acc);
    }
#pragma unroll
    for (int r = 0; r < 8; ++r) {
      const int s  = q0 + r + ((lane >> 4) << 3);
      const int dc = hh * HD_ + wid * 16 + (lane & 15);
      attnOut[((size_t)bi * S_ + s) * D_ + dc] = (_Float16)acc[r];
    }
  }
}

// ---------------------------------------------------------------------------
// K4/K5: GEMM out = A(f16) @ W(f16)^T + bias, TDM double-buffered.
// mode 0: write f16.  mode 1: outF = acc + bias + (float)addSrc.
// ---------------------------------------------------------------------------
__global__ void __launch_bounds__(256) gemm_kernel(
    const _Float16* __restrict__ A, const _Float16* __restrict__ Wh,
    const float* __restrict__ bias,
    _Float16* __restrict__ outH, float* __restrict__ outF,
    const _Float16* __restrict__ addSrc, int mode)
{
  __shared__ _Float16 At[2][128 * 40];
  __shared__ _Float16 Bt[2][64 * 40];
  const int tid  = threadIdx.x;
  const int lane = tid & 31;
  const int wid  = tid >> 5;
  const int wm   = wid >> 1, wn = wid & 1;
  const int m0   = blockIdx.x * 128;
  const int n0   = blockIdx.y * 64;

  v8f acc[4];
#pragma unroll
  for (int t = 0; t < 4; ++t) acc[t] = (v8f){};

#if HAVE_TDM
  if (wid == 0) {
    tdm_load_tile_f16(A  + (size_t)m0 * D_, lds_off(&At[0][0]), 128, B_ * S_);
    tdm_load_tile_f16(Wh + (size_t)n0 * D_, lds_off(&Bt[0][0]), 64, D_);
  }
  for (int kk = 0; kk < 16; ++kk) {
    const int bs = kk & 1;
    if (wid == 0) {
      if (kk < 15) {
        const int k1 = (kk + 1) * 32;
        tdm_load_tile_f16(A  + (size_t)m0 * D_ + k1, lds_off(&At[bs ^ 1][0]), 128, B_ * S_);
        tdm_load_tile_f16(Wh + (size_t)n0 * D_ + k1, lds_off(&Bt[bs ^ 1][0]), 64, D_);
        __builtin_amdgcn_s_wait_tensorcnt(2);
      } else {
        __builtin_amdgcn_s_wait_tensorcnt(0);
      }
    }
    __syncthreads();
    const v16h a0  = load_frag_a16(&At[bs][(wm * 32) * 40], 40);
    const v16h a1  = load_frag_a16(&At[bs][(wm * 32 + 16) * 40], 40);
    const v16h b0  = load_frag_b16(&Bt[bs][(wn * 32) * 40], 40);
    const v16h b1f = load_frag_b16(&Bt[bs][(wn * 32 + 16) * 40], 40);
    acc[0] = wmma16(a0, b0,  acc[0]);
    acc[1] = wmma16(a0, b1f, acc[1]);
    acc[2] = wmma16(a1, b0,  acc[2]);
    acc[3] = wmma16(a1, b1f, acc[3]);
    __syncthreads();
  }
#else
  for (int k0 = 0; k0 < D_; k0 += 32) {
    __syncthreads();
#pragma unroll
    for (int i = 0; i < 2; ++i) {
      const int idx = tid + i * 256;
      const int r = idx >> 2, c = (idx & 3) * 8;
      *(uint4*)&At[0][r * 40 + c] = *(const uint4*)(A + (size_t)(m0 + r) * D_ + k0 + c);
    }
    {
      const int r = tid >> 2, c = (tid & 3) * 8;
      *(uint4*)&Bt[0][r * 40 + c] = *(const uint4*)(Wh + (size_t)(n0 + r) * D_ + k0 + c);
    }
    __syncthreads();
    const v16h a0  = load_frag_a16(&At[0][(wm * 32) * 40], 40);
    const v16h a1  = load_frag_a16(&At[0][(wm * 32 + 16) * 40], 40);
    const v16h b0  = load_frag_b16(&Bt[0][(wn * 32) * 40], 40);
    const v16h b1f = load_frag_b16(&Bt[0][(wn * 32 + 16) * 40], 40);
    acc[0] = wmma16(a0, b0,  acc[0]);
    acc[1] = wmma16(a0, b1f, acc[1]);
    acc[2] = wmma16(a1, b0,  acc[2]);
    acc[3] = wmma16(a1, b1f, acc[3]);
  }
#endif

#pragma unroll
  for (int i = 0; i < 2; ++i) {
#pragma unroll
    for (int j = 0; j < 2; ++j) {
      const int n = n0 + wn * 32 + j * 16 + (lane & 15);
      const float bb = bias[n];
      const v8f a = acc[i * 2 + j];
#pragma unroll
      for (int r = 0; r < 8; ++r) {
        const int m = m0 + wm * 32 + i * 16 + r + ((lane >> 4) << 3);
        const float val = a[r] + bb;
        if (mode == 0) outH[(size_t)m * D_ + n] = (_Float16)val;
        else           outF[(size_t)m * D_ + n] = val + (float)addSrc[(size_t)m * D_ + n];
      }
    }
  }
}

// ---------------------------------------------------------------------------
// K6: LayerNorm over D=512, wave per row.
// ---------------------------------------------------------------------------
__global__ void __launch_bounds__(256) ln_kernel(
    const float* __restrict__ X, const float* __restrict__ g,
    const float* __restrict__ be, float* __restrict__ Out)
{
  const int lane = threadIdx.x & 31;
  const int wid  = threadIdx.x >> 5;
  const int m    = blockIdx.x * 8 + wid;
  const float* xr = X + (size_t)m * D_;
  float s = 0.0f, ss = 0.0f;
  float xs[16];
#pragma unroll
  for (int t = 0; t < 16; ++t) {
    const float x = xr[lane + t * 32];
    xs[t] = x; s += x; ss += x * x;
  }
#pragma unroll
  for (int o = 16; o >= 1; o >>= 1) { s += __shfl_xor(s, o, 32); ss += __shfl_xor(ss, o, 32); }
  const float mu   = s * (1.0f / 512.0f);
  const float var  = ss * (1.0f / 512.0f) - mu * mu;
  const float rstd = rsqrtf(var + 1e-5f);
#pragma unroll
  for (int t = 0; t < 16; ++t) {
    const int c = lane + t * 32;
    Out[(size_t)m * D_ + c] = (xs[t] - mu) * rstd * g[c] + be[c];
  }
}

// ---------------------------------------------------------------------------
extern "C" void kernel_launch(void* const* d_in, const int* in_sizes, int n_in,
                              void* d_out, int out_size, void* d_ws, size_t ws_size,
                              hipStream_t stream)
{
  (void)in_sizes; (void)n_in; (void)out_size; (void)ws_size;
  const float* inputs = (const float*)d_in[1];
  const float* ren    = (const float*)d_in[2];
  const float* ts     = (const float*)d_in[3];
  const float* Wq = (const float*)d_in[5];  const float* bq = (const float*)d_in[6];
  const float* Wk = (const float*)d_in[7];  const float* bk = (const float*)d_in[8];
  const float* Wv = (const float*)d_in[9];  const float* bv = (const float*)d_in[10];
  const float* Wo = (const float*)d_in[11]; const float* bo = (const float*)d_in[12];
  const float* W1 = (const float*)d_in[13]; const float* b1 = (const float*)d_in[14];
  const float* lng = (const float*)d_in[15];
  const float* lnb = (const float*)d_in[16];
  const float* l1p = (const float*)d_in[17];

  char* ws = (char*)d_ws;
  const size_t SZH  = (size_t)B_ * H_ * S_ * HD_ * 2;  // 8 MiB
  const size_t SZW  = (size_t)D_ * D_ * 2;             // 512 KiB
  _Float16* Qf   = (_Float16*)(ws);
  _Float16* Kf   = (_Float16*)(ws + SZH);
  _Float16* VT   = (_Float16*)(ws + 2 * SZH);
  float*    Mem  = (float*)   (ws + 3 * SZH);
  _Float16* attO = (_Float16*)(ws + 3 * SZH + (size_t)B_ * S_ * S_ * 4);
  _Float16* out1 = (_Float16*)((char*)attO + SZH);
  float*    out2 = (float*)   ((char*)out1 + SZH);
  _Float16* Xh   = (_Float16*)((char*)out2 + (size_t)B_ * S_ * D_ * 4);
  _Float16* Whq  = (_Float16*)((char*)Xh + SZH);
  _Float16* Whk  = (_Float16*)((char*)Whq + SZW);
  _Float16* Whv  = (_Float16*)((char*)Whk + SZW);
  _Float16* Who  = (_Float16*)((char*)Whv + SZW);
  _Float16* Wh1  = (_Float16*)((char*)Who + SZW);

  const int n4x = B_ * S_ * D_ / 4;   // 1,048,576
  const int n4w = D_ * D_ / 4;        // 65,536
  cvt_kernel<<<dim3(n4x / 256), 256, 0, stream>>>(inputs, Xh, n4x);
  cvt_kernel<<<dim3(n4w / 256), 256, 0, stream>>>(Wq, Whq, n4w);
  cvt_kernel<<<dim3(n4w / 256), 256, 0, stream>>>(Wk, Whk, n4w);
  cvt_kernel<<<dim3(n4w / 256), 256, 0, stream>>>(Wv, Whv, n4w);
  cvt_kernel<<<dim3(n4w / 256), 256, 0, stream>>>(Wo, Who, n4w);
  cvt_kernel<<<dim3(n4w / 256), 256, 0, stream>>>(W1, Wh1, n4w);

  mem_attn_kernel<<<dim3((B_ * S_) / 8), 256, 0, stream>>>(ren, ts, Mem);
  qkv_proj_kernel<<<dim3((B_ * S_) / 128, D_ / 64), 256, 0, stream>>>(
      Xh, Whq, Whk, Whv, bq, bk, bv, Qf, Kf, VT);
  attn_kernel<<<dim3(S_ / 16, B_ * H_), 128, 0, stream>>>(Qf, Kf, VT, Mem, l1p, attO);
  gemm_kernel<<<dim3((B_ * S_) / 128, D_ / 64), 256, 0, stream>>>(
      attO, Who, bo, out1, nullptr, nullptr, 0);
  gemm_kernel<<<dim3((B_ * S_) / 128, D_ / 64), 256, 0, stream>>>(
      out1, Wh1, b1, nullptr, out2, out1, 1);
  ln_kernel<<<dim3((B_ * S_) / 8), 256, 0, stream>>>(out2, lng, lnb, (float*)d_out);
}
